// HelixAffineLinear_46514495815712
// MI455X (gfx1250) — matrix operand order repaired and checked by reading, not compiled, not run on hardware
//
#include <hip/hip_runtime.h>

// ---------------------------------------------------------------------------
// HelixAffineLinear on MI455X (gfx1250, wave32):
//   out[n,o] = sum_i x[n,i]*((idx[o,i]-128)/127 * s[o,i/256] + b[o,i/256]) + bias[o]
//
// WMMA f16 GEMM, 369 GFLOP, compute-bound; HBM floor ~430MB (~18us @23.3TB/s);
// idx(180MB)+x(67MB) are L2-resident in 192MB L2.
// CDNA5 paths: global_load_async_to_lds_b128 + s_wait_asynccnt double-buffered
// prefetch of the dominant int32 index stream; v_wmma_f32_16x16x32_f16 compute.
// Wave tile 64x64 (4x4 fragments) -> 1.0 ds_load_b128 per WMMA (LDS bw ~50%).
// ---------------------------------------------------------------------------

typedef __attribute__((ext_vector_type(16))) _Float16 v16h;
typedef __attribute__((ext_vector_type(8)))  _Float16 v8h;
typedef __attribute__((ext_vector_type(4)))  _Float16 h4;
typedef __attribute__((ext_vector_type(8)))  float    v8f;
typedef __attribute__((ext_vector_type(4)))  float    f32x4;
typedef __attribute__((ext_vector_type(4)))  int      i32x4;

#define IN_F   4096
#define OUT_F  11008
#define BPR    16      // quant blocks per weight row (4096/256)

#define BM     128
#define BN     256
#define BK     64
#define PITCH  72      // LDS f16 row pitch (64 + 8 pad) to break bank conflicts

union AFrag { v16h v; v8h h[2]; };

__global__ __launch_bounds__(256)
void helix_affine_gemm(const float* __restrict__ x,
                       const int*   __restrict__ indices,
                       const float* __restrict__ scales,
                       const float* __restrict__ babs,
                       const float* __restrict__ bias,
                       float* __restrict__ out)
{
    // raw int32 index tiles, double-buffered, filled by async-to-LDS DMA
    __shared__ __align__(16) int      RawW[2][BN * BK];      // 2 x 64 KB
    __shared__ __align__(16) _Float16 Xs[BM * PITCH];        // 18 KB
    __shared__ __align__(16) _Float16 Ws[BN * PITCH];        // 36 KB

    const int tid   = threadIdx.x;
    const int lane  = tid & 31;
    const int wave  = tid >> 5;            // 0..7
    const int waveM = (wave & 1)  * 64;    // 2 waves along M
    const int waveN = (wave >> 1) * 64;    // 4 waves along N
    const int blockN = blockIdx.x * BN;
    const int blockM = blockIdx.y * BM;

    // cooperative-load coords: slot i -> tile row (tid>>4)+16*i, col4 (tid&15)
    const int ldRow = tid >> 4;            // 0..15
    const int ldC4  = tid & 15;            // 0..15 (x4 elements)

    v8f acc[4][4] = {{{0.f}}};             // [mfrag][nfrag]

    const int arow = waveM + (lane & 15);
    const int ak   = (lane >> 4) * 8;
    const int bcol = waveN + (lane & 15);
    const int bk   = (lane >> 4) * 8;

    // ---- async prefetch of a W index tile (16 x b128 per thread) ----
    auto issueW = [&](int ktn, int pn) {
        #pragma unroll
        for (int i = 0; i < 16; ++i) {
            const int row = ldRow + 16 * i;          // 0..255
            const unsigned lds = (unsigned)(uintptr_t)
                &RawW[pn][row * BK + ldC4 * 4];
            const unsigned long long ga = (unsigned long long)(uintptr_t)
                (indices + (size_t)(blockN + row) * IN_F + ktn + ldC4 * 4);
            asm volatile("global_load_async_to_lds_b128 %0, %1, off"
                         :: "v"(lds), "v"(ga) : "memory");
        }
    };

    // ---- register prefetch of an X tile (8 x float4 per thread) ----
    f32x4 xr[8];
    auto loadX = [&](int ktn) {
        #pragma unroll
        for (int i = 0; i < 8; ++i)
            xr[i] = *(const f32x4*)(x + (size_t)(blockM + ldRow + 16 * i) * IN_F
                                      + ktn + ldC4 * 4);
    };

    issueW(0, 0);
    loadX(0);

    int p = 0;
    for (int kt = 0; kt < IN_F; kt += BK) {
        const int  qb   = kt >> 8;               // quant block, const per chunk
        const bool more = (kt + BK) < IN_F;

        // prefetch next chunk's indices, then wait for current chunk's DMA
        if (more) {
            issueW(kt + BK, p ^ 1);
            asm volatile("s_wait_asynccnt 0x10" ::: "memory");
        } else {
            asm volatile("s_wait_asynccnt 0x0" ::: "memory");
        }
        __syncthreads();   // RawW[p] visible everywhere; f16 tiles free again

        // ---- convert X regs -> f16 LDS ----
        #pragma unroll
        for (int i = 0; i < 8; ++i) {
            const int row = ldRow + 16 * i;
            h4 hv;
            hv.x = (_Float16)xr[i].x; hv.y = (_Float16)xr[i].y;
            hv.z = (_Float16)xr[i].z; hv.w = (_Float16)xr[i].w;
            *(h4*)(&Xs[row * PITCH + ldC4 * 4]) = hv;
        }

        // ---- dequant RawW[p] -> f16 LDS (256 rows) ----
        #pragma unroll
        for (int i = 0; i < 16; ++i) {
            const int row = ldRow + 16 * i;
            const int o   = blockN + row;
            const float s127 = scales[o * BPR + qb] * (1.0f / 127.0f);
            const float ab   = babs[o * BPR + qb];
            i32x4 q = *(const i32x4*)(&RawW[p][row * BK + ldC4 * 4]);
            h4 hv;
            hv.x = (_Float16)((float)(q.x - 128) * s127 + ab);
            hv.y = (_Float16)((float)(q.y - 128) * s127 + ab);
            hv.z = (_Float16)((float)(q.z - 128) * s127 + ab);
            hv.w = (_Float16)((float)(q.w - 128) * s127 + ab);
            *(h4*)(&Ws[row * PITCH + ldC4 * 4]) = hv;
        }

        __syncthreads();   // f16 tiles ready

        // issue next X tile loads; consumed only after the next barrier
        if (more) loadX(kt + BK);

        // ---- 2 K-steps of 32: 32 x v_wmma_f32_16x16x32_f16 per chunk ----
        #pragma unroll
        for (int ks = 0; ks < BK; ks += 32) {
            AFrag a[4];
            #pragma unroll
            for (int mf = 0; mf < 4; ++mf) {
                const _Float16* pa = &Xs[(arow + mf * 16) * PITCH + ks + ak];
                a[mf].h[0] = *(const v8h*)pa;
                a[mf].h[1] = *(const v8h*)(pa + 16);
            }
            #pragma unroll
            for (int nf = 0; nf < 4; ++nf) {
                AFrag b;
                const _Float16* pb = &Ws[(bcol + nf * 16) * PITCH + ks + bk];
                b.h[0] = *(const v8h*)pb;
                b.h[1] = *(const v8h*)(pb + 16);
                #pragma unroll
                for (int mf = 0; mf < 4; ++mf)
                    acc[mf][nf] = __builtin_amdgcn_wmma_f32_16x16x32_f16(
                        false, a[mf].v, false, b.v,
                        (short)0, acc[mf][nf], false, false);
            }
        }

        p ^= 1;
    }

    // ---- epilogue: + bias[o]; C layout VGPR r -> M=r / M=8+r per half-wave
    const int crow0 = blockM + waveM + 8 * (lane >> 4);
    const int ccol0 = blockN + waveN + (lane & 15);
    #pragma unroll
    for (int nf = 0; nf < 4; ++nf) {
        const int   col = ccol0 + nf * 16;
        const float bv  = bias[col];
        #pragma unroll
        for (int mf = 0; mf < 4; ++mf) {
            #pragma unroll
            for (int r = 0; r < 8; ++r) {
                const int m = crow0 + mf * 16 + r;
                out[(size_t)m * OUT_F + col] = acc[mf][nf][r] + bv;
            }
        }
    }
}

extern "C" void kernel_launch(void* const* d_in, const int* in_sizes, int n_in,
                              void* d_out, int out_size, void* d_ws, size_t ws_size,
                              hipStream_t stream) {
    const float* xp  = (const float*)d_in[0];   // [2,2048,4096] f32
    const int*   idx = (const int*)d_in[1];     // [11008,4096] i32
    const float* sc  = (const float*)d_in[2];   // [11008*16] f32
    const float* ab  = (const float*)d_in[3];   // [11008*16] f32
    const float* bs  = (const float*)d_in[4];   // [11008] f32
    float* out = (float*)d_out;

    const int Mrows = in_sizes[0] / IN_F;       // 4096
    dim3 grid(OUT_F / BN, Mrows / BM);          // (43, 32)
    helix_affine_gemm<<<grid, 256, 0, stream>>>(xp, idx, sc, ab, bs, out);
}
